// HardnessAwareKernelizedSupCon_14156212207653
// MI455X (gfx1250) — matrix operand order repaired and checked by reading, not compiled, not run on hardware
//
#include <hip/hip_runtime.h>
#include <hip/hip_bf16.h>

typedef __attribute__((ext_vector_type(16))) _Float16 v16h;
typedef __attribute__((ext_vector_type(8)))  _Float16 v8h;
typedef __attribute__((ext_vector_type(4)))  _Float16 v4h;
typedef __attribute__((ext_vector_type(8)))  float    v8f;
typedef __attribute__((ext_vector_type(4)))  float    v4f;
typedef __attribute__((ext_vector_type(4)))  unsigned int u32x4;
typedef __attribute__((ext_vector_type(8)))  int      i32x8;
typedef __attribute__((ext_vector_type(4)))  int      i32x4;

#define BSZ     4096
#define NVIEWS  2
#define DIM     128
#define N_TOT   (BSZ*NVIEWS)     // 8192
#define TILE    16
#define NTILES  (N_TOT/TILE)     // 512
#define CB      64               // columns per LDS panel
#define NPANELS (N_TOT/CB)       // 128
#define PANEL_HALVES (CB*DIM)    // 8192 f16 = 16 KB
#define PANEL_U8     (CB*DIM*2/8) // 2048 8-byte units
#define INV_T   (1.0f/0.07f)
#define EPSV    1e-8f

// ---------------------------------------------------------------------------
// Kernel 1: normalize each feature row, stack view-major, convert to f16.
// One wave per stacked row; 4 floats per lane; wave-reduce sum of squares.
// ---------------------------------------------------------------------------
__global__ void __launch_bounds__(32)
supcon_prep(const float* __restrict__ feat, const float* __restrict__ lab,
            _Float16* __restrict__ fh, float* __restrict__ y)
{
    const int k    = blockIdx.x;        // stacked row (k = v*BSZ + i)
    const int i    = k & (BSZ - 1);
    const int v    = k >> 12;
    const int lane = threadIdx.x;

    const float* src = feat + ((size_t)i * NVIEWS + v) * DIM + lane * 4;
    v4f x = *(const v4f*)src;
    float ssq = x.x*x.x + x.y*x.y + x.z*x.z + x.w*x.w;
    #pragma unroll
    for (int off = 16; off > 0; off >>= 1)
        ssq += __shfl_xor(ssq, off, 32);
    const float rn = rsqrtf(ssq + 1e-12f);

    v4h h;
    h.x = (_Float16)(x.x * rn);
    h.y = (_Float16)(x.y * rn);
    h.z = (_Float16)(x.z * rn);
    h.w = (_Float16)(x.w * rn);
    *(v4h*)(fh + (size_t)k * DIM + lane * 4) = h;

    if (lane == 0) y[k] = lab[i];
}

// ---------------------------------------------------------------------------
// TDM: issue an async 1-D tensor_load_to_lds of one contiguous column panel
// (CB rows x 128 K of f16 = PANEL_U8 8-byte elements). D# per ISA §8.3-8.4.
// 6-arg builtin form (clang-23): (u32x4 g0, i32x8 g1, i32x4, i32x4, i32x8, cpol)
// ---------------------------------------------------------------------------
__device__ __forceinline__ void tdm_load_panel(const _Float16* gsrc,
                                               const _Float16* lds_dst)
{
    const unsigned long long ga = (unsigned long long)(uintptr_t)gsrc;
    // flat LDS pointer: addr[31:0] is the LDS byte offset (ISA aperture rule)
    const unsigned lds_off = (unsigned)(uintptr_t)lds_dst;

    u32x4 g0;
    g0[0] = 1u;                                   // count=1, user descriptor
    g0[1] = lds_off;                              // lds_addr (bytes)
    g0[2] = (unsigned)(ga & 0xffffffffu);         // global_addr[31:0]
    g0[3] = (unsigned)((ga >> 32) & 0x01ffffffu)  // global_addr[56:32]
          | (2u << 30);                           // type = 2 ("image")

    i32x8 g1;
    g1[0] = (3 << 16);                            // data_size=3 (8B); mask=0
    g1[1] = (int)((PANEL_U8 & 0xffff) << 16);     // tensor_dim0[15:0]
    g1[2] = (int)((PANEL_U8 >> 16) & 0xffff);     // tensor_dim0[31:16]
    g1[3] = (int)((PANEL_U8 & 0xffff) << 16);     // tile_dim0
    g1[4] = 0;                                    // tile_dim1/2 unused
    g1[5] = (int)PANEL_U8;                        // tensor_dim0_stride[31:0]
    g1[6] = 0;
    g1[7] = 0;

    i32x4 z4 = {0, 0, 0, 0};                      // groups 2/3 unused (<=2D)
    i32x8 z8 = {0, 0, 0, 0, 0, 0, 0, 0};
    __builtin_amdgcn_tensor_load_to_lds(g0, g1, z4, z4, z8, 0);
}

// ---------------------------------------------------------------------------
// Kernel 2: fused Gram + kernel-weighted log-prob statistics.
// 256 threads = 8 waves; each wave owns 16 rows. Column panels are staged in
// LDS by the Tensor Data Mover (double-buffered), B fragments read from LDS.
// z accumulates exp(s - 1/T) against the fixed bound s<=1/T (cos<=1), so the
// reference logsumexp is recovered exactly as z*exp(1/T - max) at the end.
// ---------------------------------------------------------------------------
__global__ void __launch_bounds__(256)
supcon_main(const _Float16* __restrict__ fh, const float* __restrict__ y,
            float* __restrict__ row_val)
{
    __shared__ _Float16 smem[2][PANEL_HALVES];    // 2 x 16 KB panels

    const int lane    = threadIdx.x & 31;
    const int wave    = threadIdx.x >> 5;
    const int rowbase = (blockIdx.x * 8 + wave) * TILE;
    const int n       = lane & 15;
    const int hi      = lane >> 4;

    // --- A fragments: 16 rows x 128 K, lane half hi holds K=c*32+hi*16..+15 --
    const int arow = rowbase + n;
    v16h afrag[4];
    #pragma unroll
    for (int c = 0; c < 4; ++c) {
        const _Float16* pr = fh + (size_t)arow * DIM + c * 32 + hi * 16;
        afrag[c] = *(const v16h*)(pr);
    }

    float yr[8];
    #pragma unroll
    for (int r = 0; r < 8; ++r) yr[r] = y[rowbase + hi * 8 + r];

    float mrow[8], zrow[8], asum[8], wsum[8];
    #pragma unroll
    for (int r = 0; r < 8; ++r) { mrow[r] = -3.0e38f; zrow[r] = 0.f; asum[r] = 0.f; wsum[r] = 0.f; }

    // prime the pipeline: panel 0 into buffer 0
    if (threadIdx.x < 32) tdm_load_panel(fh, &smem[0][0]);

    for (int p = 0; p < NPANELS; ++p) {
        const _Float16* buf = &smem[p & 1][0];

        if (threadIdx.x < 32) {
            if (p + 1 < NPANELS) {
                // issue next panel, then wait until panel p landed (in-order)
                tdm_load_panel(fh + (size_t)(p + 1) * PANEL_HALVES,
                               &smem[(p + 1) & 1][0]);
                __builtin_amdgcn_s_wait_tensorcnt((short)1);
            } else {
                __builtin_amdgcn_s_wait_tensorcnt((short)0);
            }
        }
        __syncthreads();                          // panel p visible to all waves

        #pragma unroll
        for (int t = 0; t < CB / TILE; ++t) {     // 4 tiles per panel
            const int jt        = p * (CB / TILE) + t;
            const int col       = jt * TILE + n;  // global column
            const int col_local = t * TILE + n;   // column within LDS panel
            const float ycol = y[col];

            // B fragment from LDS: lane half hi holds K=c*32+hi*16..+15
            v8f acc = {};
            #pragma unroll
            for (int c = 0; c < 4; ++c) {
                const _Float16* pb = buf + (size_t)col_local * DIM + c * 32 + hi * 16;
                v16h b = *(const v16h*)pb;
                acc = __builtin_amdgcn_wmma_f32_16x16x32_f16(
                          false, afrag[c], false, b, (short)0, acc, false, false);
            }

            #pragma unroll
            for (int r = 0; r < 8; ++r) {
                const int grow = rowbase + hi * 8 + r;
                const float s   = acc[r] * INV_T;
                const float dy  = yr[r] - ycol;
                const float wij = __expf(-0.5f * dy * dy);
                wsum[r] += wij;
                asum[r]  = fmaf(wij, s, asum[r]);
                mrow[r]  = fmaxf(mrow[r], s);
                const float e2 = (grow == col) ? 0.f : __expf(s - INV_T);
                zrow[r] += e2;
            }
        }
        __syncthreads();                          // done reading buf before reuse
    }

    // --- merge partial stats across the 16 lanes sharing each row ----------
    #pragma unroll
    for (int off = 1; off < 16; off <<= 1) {
        #pragma unroll
        for (int r = 0; r < 8; ++r) {
            mrow[r] = fmaxf(mrow[r], __shfl_xor(mrow[r], off, 32));
            zrow[r] += __shfl_xor(zrow[r], off, 32);
            asum[r] += __shfl_xor(asum[r], off, 32);
            wsum[r] += __shfl_xor(wsum[r], off, 32);
        }
    }

    if (n == 0) {
        #pragma unroll
        for (int r = 0; r < 8; ++r) {
            const float Zref = zrow[r] * __expf(INV_T - mrow[r]);
            const float W    = wsum[r];
            const float val  = (asum[r] - W * (mrow[r] + __logf(Zref + EPSV)))
                               / fmaxf(W, EPSV);
            row_val[rowbase + hi * 8 + r] = val;
        }
    }
}

// ---------------------------------------------------------------------------
// Kernel 3: deterministic reduction of per-row values -> scalar loss.
// ---------------------------------------------------------------------------
__global__ void __launch_bounds__(256)
supcon_finalize(const float* __restrict__ row_val, float* __restrict__ out)
{
    __shared__ float sm[256];
    float s = 0.f;
    for (int i = threadIdx.x; i < N_TOT; i += 256) s += row_val[i];
    sm[threadIdx.x] = s;
    __syncthreads();
    #pragma unroll
    for (int st = 128; st > 0; st >>= 1) {
        if (threadIdx.x < st) sm[threadIdx.x] += sm[threadIdx.x + st];
        __syncthreads();
    }
    if (threadIdx.x == 0) out[0] = -sm[0] / (float)N_TOT;  // T/BASE_T == 1
}

// ---------------------------------------------------------------------------
extern "C" void kernel_launch(void* const* d_in, const int* in_sizes, int n_in,
                              void* d_out, int out_size, void* d_ws, size_t ws_size,
                              hipStream_t stream)
{
    const float* features = (const float*)d_in[0];   // [4096, 2, 128] f32
    const float* labels   = (const float*)d_in[1];   // [4096] f32
    float*       out      = (float*)d_out;           // scalar f32

    char* ws = (char*)d_ws;
    _Float16* fh      = (_Float16*)(ws);                          // 2 MB
    float*    y       = (float*)(ws + (size_t)N_TOT * DIM * 2);   // 32 KB
    float*    row_val = y + N_TOT;                                // 32 KB

    supcon_prep<<<N_TOT, 32, 0, stream>>>(features, labels, fh, y);
    supcon_main<<<N_TOT / (8 * TILE), 256, 0, stream>>>(fh, y, row_val);
    supcon_finalize<<<1, 256, 0, stream>>>(row_val, out);
}
